// SGFormerAttention_63445256896637
// MI455X (gfx1250) — compile-verified, matrix operand order checked by reading, not run on hardware
//
#include <hip/hip_runtime.h>

// ---------------------------------------------------------------------------
// SGFormer linear attention for MI455X (gfx1250, wave32, WMMA bf16 16x16x32)
//
// N=131072 nodes, C=512, M=128 head ch, B=16 graphs, L=8192 (reshape).
// Un-normalized math; global Frobenius norms folded into one scalar
// inv_qk = 1/(||Q||*||K||) applied in the epilogue.
//
// Data-movement design:
//  - A operands: natural [row][k] tiles -> 2x ds_load_b128 per fragment.
//  - B operands: FRAGMENT-MAJOR layout (lane's 16 bf16 contiguous per tile),
//    produced at the source (Wsw / Vsw / kvsw) -> 2x ds_load_b128, 0 VALU.
//  - K stored transposed Kt[g][m][l]; all GEMM loops double-buffered in LDS
//    (1 barrier per K-step, global loads overlap WMMA).
//  - All f32->bf16 conversion via vector fptrunc -> pure v_cvt_pk_bf16_f32.
// ---------------------------------------------------------------------------

typedef __attribute__((ext_vector_type(16))) __bf16 v16bf;
typedef __attribute__((ext_vector_type(8)))  __bf16 v8bf;
typedef __attribute__((ext_vector_type(16))) float  v16f;
typedef __attribute__((ext_vector_type(8)))  float  v8f;

#define N_NODES 131072
#define CH      512
#define HCH     128
#define NGRAPH  16
#define LNODES  8192

#define LDA  40   // LDS element stride for A tiles [rows][32] bf16
#define RSTR 24   // LDS elements per B record (16 data + 8 pad = 48 B)

#define WMMA_BF16(a, b, c) \
  __builtin_amdgcn_wmma_f32_16x16x32_bf16(false, (a), false, (b), (short)0, \
                                          (c), false, false)

static __device__ inline v8f zero8() {
  v8f z;
#pragma unroll
  for (int e = 0; e < 8; ++e) z[e] = 0.0f;
  return z;
}

// v8f accumulator -> 8 packed bf16 (uint4) via vector fptrunc (4x cvt_pk)
static __device__ inline uint4 cvt8_bf16(v8f a) {
  union { v8bf v; uint4 q; } t;
  t.v = __builtin_convertvector(a, v8bf);
  return t.q;
}

static __device__ inline float unpk_lo(unsigned w) {
  return (float)__builtin_bit_cast(__bf16, (unsigned short)(w & 0xffffu));
}
static __device__ inline float unpk_hi(unsigned w) {
  return (float)__builtin_bit_cast(__bf16, (unsigned short)(w >> 16));
}

// A operand: 16x32 bf16 tile from LDS [row][k], per CDNA5 ISA 7.12.2.
static __device__ inline v16bf load_frag_a(const __bf16* s, int row0, int ld) {
  const int lane = threadIdx.x & 31;
  const int r  = lane & 15;
  const int kh = (lane >> 4) << 3;
  const __bf16* p = s + (row0 + r) * ld;
  v16bf a;
#pragma unroll
  for (int e = 0; e < 16; ++e) {
    const int pr = e >> 1, w = e & 1;
    const int k = ((pr >= 4) ? 16 : 0) + kh + ((pr & 3) << 1) + w;
    a[e] = p[k];
  }
  return a;
}

// B operand from fragment-major LDS: 2x ds_load_b128, no packing.
static __device__ inline v16bf load_frag_b_fm(const __bf16* s, int tile) {
  const int lane = threadIdx.x & 31;
  const uint4* p =
      reinterpret_cast<const uint4*>(s + (tile * 32 + lane) * RSTR);
  union { v16bf v; uint4 q[2]; } f;
  f.q[0] = p[0];
  f.q[1] = p[1];
  return f.v;
}

// copy 32 B global -> LDS
static __device__ inline void cp32b(__bf16* dst, const __bf16* src) {
  const uint4* s4 = reinterpret_cast<const uint4*>(src);
  uint4* d4 = reinterpret_cast<uint4*>(dst);
  d4[0] = s4[0];
  d4[1] = s4[1];
}
static __device__ inline void st32b(__bf16* dst, const uint4* r) {
  uint4* d4 = reinterpret_cast<uint4*>(dst);
  d4[0] = r[0];
  d4[1] = r[1];
}
static __device__ inline void ld32b(uint4* r, const __bf16* src) {
  const uint4* s4 = reinterpret_cast<const uint4*>(src);
  r[0] = s4[0];
  r[1] = s4[1];
}

// 16 floats (in 4 float4 regs) -> 16 bf16 -> 2 uint4 LDS stores
static __device__ inline void cvt_store_x(__bf16* dst, const float4* xf) {
  union { float4 f4[4]; v16f v; } u;
#pragma unroll
  for (int j2 = 0; j2 < 4; ++j2) u.f4[j2] = xf[j2];
  union { v16bf v; uint4 q[2]; } xb;
  xb.v = __builtin_convertvector(u.v, v16bf);
  uint4* d4 = reinterpret_cast<uint4*>(dst);
  d4[0] = xb.q[0];
  d4[1] = xb.q[1];
}

// ---------------------------------------------------------------------------
__global__ void zero_kernel(float* __restrict__ p, int n) {
  int i = blockIdx.x * blockDim.x + threadIdx.x;
  if (i < n) p[i] = 0.0f;
}

// W -> fragment-major Wsw (per which; k-dim = c, n-dim = m)
__global__ void wt_kernel(const float* __restrict__ Wq,
                          const float* __restrict__ Wk,
                          const float* __restrict__ Wv,
                          __bf16* __restrict__ wsw) {
  int idx = blockIdx.x * blockDim.x + threadIdx.x;
  if (idx >= 3 * CH * HCH) return;
  const int which = idx >> 16;
  const int r = idx & 65535;
  const int c = r >> 7, m = r & 127;
  const int cc = c >> 5, k = c & 31;
  const int kh = (k >> 3) & 1;
  const int ev = 2 * (((k >= 16) ? 4 : 0) + ((k & 7) >> 1)) + (k & 1);
  const int rec = (cc * 8 + (m >> 4)) * 32 + (m & 15) + 16 * kh;
  const float* W = (which == 0) ? Wq : (which == 1) ? Wk : Wv;
  wsw[(size_t)which * 65536 + rec * 16 + ev] = (__bf16)W[m * CH + c];
}

// ---------------------------------------------------------------------------
// Projection: Y = X @ W^T (blockIdx.y in {q,k,v}); 8 waves, 128x128 tile.
__global__ __launch_bounds__(256) void proj_kernel(
    const float* __restrict__ x, const __bf16* __restrict__ wsw,
    __bf16* __restrict__ qb, __bf16* __restrict__ kt, __bf16* __restrict__ vsw,
    float* __restrict__ sumsq, float* __restrict__ ksum) {
  __shared__ __bf16 Xs[2][128 * LDA];
  __shared__ __bf16 Ws[2][256 * RSTR];

  const int which = blockIdx.y;
  const int n0 = blockIdx.x * 128;
  const int g = n0 >> 13;
  const __bf16* wtb = wsw + which * (CH * HCH);

  const int tid = threadIdx.x;
  const int wid = tid >> 5, lane = tid & 31;
  const int wr = wid & 3, wc = wid >> 2;

  v8f acc[2][4];
#pragma unroll
  for (int i = 0; i < 2; ++i)
#pragma unroll
    for (int j = 0; j < 4; ++j) acc[i][j] = zero8();

  const int xrow = tid >> 1;
  const int xcg  = (tid & 1) * 16;
  const float* xbase = x + (size_t)(n0 + xrow) * CH + xcg;

  // prologue: stage chunk 0 into buffer 0
  {
    float4 xf[4];
#pragma unroll
    for (int j2 = 0; j2 < 4; ++j2)
      xf[j2] = *reinterpret_cast<const float4*>(xbase + j2 * 4);
    cvt_store_x(&Xs[0][xrow * LDA + xcg], xf);
    cp32b(&Ws[0][tid * RSTR], wtb + tid * 16);
  }
  __syncthreads();

  for (int it = 0; it < 16; ++it) {
    const int cur = it & 1;
    const bool more = (it + 1) < 16;
    float4 xf[4];
    uint4 wrec[2];
    if (more) {  // issue next-chunk global loads before the WMMAs
      const float* xp = xbase + (it + 1) * 32;
#pragma unroll
      for (int j2 = 0; j2 < 4; ++j2)
        xf[j2] = *reinterpret_cast<const float4*>(xp + j2 * 4);
      ld32b(wrec, wtb + (size_t)(it + 1) * 4096 + tid * 16);
      __builtin_prefetch(xp + 32, 0, 1);  // global_prefetch_b8 (chunk it+2)
    }

    v16bf a0 = load_frag_a(&Xs[cur][0], wr * 32,      LDA);
    v16bf a1 = load_frag_a(&Xs[cur][0], wr * 32 + 16, LDA);
#pragma unroll
    for (int j = 0; j < 4; ++j) {
      v16bf b = load_frag_b_fm(&Ws[cur][0], wc * 4 + j);
      acc[0][j] = WMMA_BF16(a0, b, acc[0][j]);
      acc[1][j] = WMMA_BF16(a1, b, acc[1][j]);
    }

    if (more) {  // stores into the other buffer after the WMMAs
      const int nxt = cur ^ 1;
      cvt_store_x(&Xs[nxt][xrow * LDA + xcg], xf);
      st32b(&Ws[nxt][tid * RSTR], wrec);
    }
    __syncthreads();
  }

  const int rh = (lane >> 4) * 8;
  const int cl = lane & 15;
  const int col0 = wc * 64;

  if (which == 0) {
    // Q natural row-major (A operand + rowdot later)
#pragma unroll
    for (int i = 0; i < 2; ++i) {
      const int rbase = n0 + wr * 32 + i * 16 + rh;
#pragma unroll
      for (int e = 0; e < 8; ++e)
#pragma unroll
        for (int j = 0; j < 4; ++j)
          qb[(size_t)(rbase + e) * HCH + col0 + j * 16 + cl] =
              (__bf16)acc[i][j][e];
  } } else if (which == 1) {
    // K transposed: Kt[(g*128+m)*8192 + l]; 8 consecutive l -> one b128 store
    const int lb0 = (n0 - g * LNODES) + wr * 32;
#pragma unroll
    for (int i = 0; i < 2; ++i) {
      const int lbase = lb0 + i * 16 + rh;
#pragma unroll
      for (int j = 0; j < 4; ++j) {
        const int m = col0 + j * 16 + cl;
        *reinterpret_cast<uint4*>(kt + (size_t)(g * HCH + m) * LNODES + lbase) =
            cvt8_bf16(acc[i][j]);
      }
    }
  } else {
    // V fragment-major per graph (k-dim = l, n-dim = d): one uint4 per tile
    const size_t vsg = (size_t)g * LNODES * HCH;
#pragma unroll
    for (int i = 0; i < 2; ++i) {
      const int l0 = (n0 - g * LNODES) + wr * 32 + i * 16 + rh;  // 8-aligned
      const int c = l0 >> 5, k0 = l0 & 31;
      const int kh = (k0 >> 3) & 1;
      const int ev0 = 2 * (((k0 >= 16) ? 4 : 0) + ((k0 & 7) >> 1));
#pragma unroll
      for (int j = 0; j < 4; ++j) {
        const int d = col0 + j * 16 + cl;
        const int rec = (c * 8 + (d >> 4)) * 32 + (d & 15) + 16 * kh;
        *reinterpret_cast<uint4*>(vsw + vsg + (size_t)rec * 16 + ev0) =
            cvt8_bf16(acc[i][j]);
      }
    }
  }

  if (which < 2) {  // Frobenius sums of squares
    float ss = 0.0f;
#pragma unroll
    for (int i = 0; i < 2; ++i)
#pragma unroll
      for (int j = 0; j < 4; ++j)
#pragma unroll
        for (int e = 0; e < 8; ++e) ss += acc[i][j][e] * acc[i][j][e];
#pragma unroll
    for (int m2 = 16; m2 >= 1; m2 >>= 1) ss += __shfl_xor(ss, m2, 32);
    if (lane == 0) atomicAdd(&sumsq[which], ss);
  }
  if (which == 1) {  // K column sums per graph
#pragma unroll
    for (int j = 0; j < 4; ++j) {
      float cs = 0.0f;
#pragma unroll
      for (int i = 0; i < 2; ++i)
#pragma unroll
        for (int e = 0; e < 8; ++e) cs += acc[i][j][e];
      cs += __shfl_xor(cs, 16, 32);
      if (lane < 16) atomicAdd(&ksum[g * HCH + col0 + j * 16 + cl], cs);
    }
  }
}

// ---------------------------------------------------------------------------
// KV: kvsw[g] = fragment-major bf16 K[g]^T V[g]. One block per graph.
__global__ __launch_bounds__(256) void kv_kernel(
    const __bf16* __restrict__ kt, const __bf16* __restrict__ vsw,
    __bf16* __restrict__ kvsw) {
  __shared__ __bf16 KT[2][128 * LDA];   // A tile [m][l-chunk], natural
  __shared__ __bf16 Vs[2][256 * RSTR];  // B tile, fragment-major verbatim
  const int g = blockIdx.x;
  const int tid = threadIdx.x, wid = tid >> 5, lane = tid & 31;
  const int mr = wid & 3, dc = wid >> 2;

  v8f acc[2][4];
#pragma unroll
  for (int i = 0; i < 2; ++i)
#pragma unroll
    for (int j = 0; j < 4; ++j) acc[i][j] = zero8();

  const int am = tid >> 1;          // 0..127 m row
  const int ah = (tid & 1) * 16;    // l half-chunk
  const size_t ktg = (size_t)g * HCH * LNODES;
  const size_t vsg = (size_t)g * LNODES * HCH;
  const __bf16* ktb = kt + ktg + (size_t)am * LNODES + ah;

  cp32b(&KT[0][am * LDA + ah], ktb);
  cp32b(&Vs[0][tid * RSTR], vsw + vsg + tid * 16);
  __syncthreads();

  for (int it = 0; it < 256; ++it) {
    const int cur = it & 1;
    const bool more = (it + 1) < 256;
    uint4 kr[2], vr[2];
    if (more) {
      ld32b(kr, ktb + (it + 1) * 32);
      ld32b(vr, vsw + vsg + (size_t)(it + 1) * 4096 + tid * 16);
    }

    v16bf a0 = load_frag_a(&KT[cur][0], mr * 32,      LDA);
    v16bf a1 = load_frag_a(&KT[cur][0], mr * 32 + 16, LDA);
#pragma unroll
    for (int j = 0; j < 4; ++j) {
      v16bf b = load_frag_b_fm(&Vs[cur][0], dc * 4 + j);
      acc[0][j] = WMMA_BF16(a0, b, acc[0][j]);
      acc[1][j] = WMMA_BF16(a1, b, acc[1][j]);
    }

    if (more) {
      const int nxt = cur ^ 1;
      st32b(&KT[nxt][am * LDA + ah], kr);
      st32b(&Vs[nxt][tid * RSTR], vr);
    }
    __syncthreads();
  }

  // store kvs fragment-major (k-dim = m, n-dim = d)
  const int rh = (lane >> 4) * 8, cl = lane & 15;
  __bf16* kvg = kvsw + (size_t)g * HCH * HCH;
#pragma unroll
  for (int i = 0; i < 2; ++i) {
    const int mbase = mr * 32 + i * 16 + rh;  // 8-aligned
    const int c = mbase >> 5, k0 = mbase & 31;
    const int kh = (k0 >> 3) & 1;
    const int ev0 = 2 * (((k0 >= 16) ? 4 : 0) + ((k0 & 7) >> 1));
#pragma unroll
    for (int j = 0; j < 4; ++j) {
      const int d = dc * 64 + j * 16 + cl;
      const int rec = (c * 8 + (d >> 4)) * 32 + (d & 15) + 16 * kh;
      *reinterpret_cast<uint4*>(kvg + (size_t)rec * 16 + ev0) =
          cvt8_bf16(acc[i][j]);
    }
  }
}

// ---------------------------------------------------------------------------
// Output: out = (Q@kvs * inv_qk + 16*V) / (Q.ksum * inv_qk + 16)
__global__ __launch_bounds__(256) void out_kernel(
    const __bf16* __restrict__ qbf, const __bf16* __restrict__ vsw,
    const __bf16* __restrict__ kvsw, const float* __restrict__ ksum,
    const float* __restrict__ sumsq, float* __restrict__ out) {
  __shared__ __bf16 Qs[2][128 * LDA];
  __shared__ __bf16 Bs[2][256 * RSTR];
  __shared__ float rowdot[128];

  const int g = blockIdx.x >> 6;
  const int n0 = blockIdx.x * 128;
  const int tid = threadIdx.x, wid = tid >> 5, lane = tid & 31;
  const int wr = wid & 3, wc = wid >> 2;

  const float inv_qk = rsqrtf(sumsq[0]) * rsqrtf(sumsq[1]);

  if (tid < 128) {  // normalizer row-dot for this block's rows (uint4 reads)
    float dsum = 0.0f;
    const uint4* qp4 =
        reinterpret_cast<const uint4*>(qbf + (size_t)(n0 + tid) * HCH);
    const float* kp = ksum + g * HCH;
#pragma unroll 4
    for (int it = 0; it < 16; ++it) {
      uint4 w = qp4[it];
      const float* k8 = kp + it * 8;
      dsum += unpk_lo(w.x) * k8[0] + unpk_hi(w.x) * k8[1];
      dsum += unpk_lo(w.y) * k8[2] + unpk_hi(w.y) * k8[3];
      dsum += unpk_lo(w.z) * k8[4] + unpk_hi(w.z) * k8[5];
      dsum += unpk_lo(w.w) * k8[6] + unpk_hi(w.w) * k8[7];
    }
    rowdot[tid] = dsum;
  }

  v8f acc[2][4];
#pragma unroll
  for (int i = 0; i < 2; ++i)
#pragma unroll
    for (int j = 0; j < 4; ++j) acc[i][j] = zero8();

  const int qrow = tid >> 1, qcg = (tid & 1) * 16;
  const __bf16* kvg = kvsw + (size_t)g * HCH * HCH;
  const size_t vsg = (size_t)g * LNODES * HCH;
  const __bf16* qpb = qbf + (size_t)(n0 + qrow) * HCH + qcg;

  cp32b(&Qs[0][qrow * LDA + qcg], qpb);
  cp32b(&Bs[0][tid * RSTR], kvg + tid * 16);
  __syncthreads();

  for (int it = 0; it < 4; ++it) {
    const int cur = it & 1;
    const bool more = (it + 1) < 4;
    uint4 qr[2], br[2];
    if (more) {
      ld32b(qr, qpb + (it + 1) * 32);
      ld32b(br, kvg + (size_t)(it + 1) * 4096 + tid * 16);
    }

    v16bf a0 = load_frag_a(&Qs[cur][0], wr * 32,      LDA);
    v16bf a1 = load_frag_a(&Qs[cur][0], wr * 32 + 16, LDA);
#pragma unroll
    for (int j = 0; j < 4; ++j) {
      v16bf b = load_frag_b_fm(&Bs[cur][0], wc * 4 + j);
      acc[0][j] = WMMA_BF16(a0, b, acc[0][j]);
      acc[1][j] = WMMA_BF16(a1, b, acc[1][j]);
    }

    if (more) {
      const int nxt = cur ^ 1;
      st32b(&Qs[nxt][qrow * LDA + qcg], qr);
      st32b(&Bs[nxt][tid * RSTR], br);
    }
    __syncthreads();
  }

  const int rh = (lane >> 4) * 8, cl = lane & 15, col0 = wc * 64;
#pragma unroll
  for (int i = 0; i < 2; ++i) {
    const int rl = wr * 32 + i * 16 + rh;              // 8-aligned
    const int lbase = n0 + rl - g * LNODES;            // local l
    const int c = lbase >> 5, k0 = lbase & 31;
    const int kh = (k0 >> 3) & 1;
    const int ev0 = 2 * (((k0 >= 16) ? 4 : 0) + ((k0 & 7) >> 1));
#pragma unroll
    for (int j = 0; j < 4; ++j) {
      const int d = col0 + j * 16 + cl;
      const int rec = (c * 8 + (d >> 4)) * 32 + (d & 15) + 16 * kh;
      const __bf16* vp = vsw + vsg + (size_t)rec * 16 + ev0;
#pragma unroll
      for (int e2 = 0; e2 < 4; ++e2) {
        const unsigned wv = *reinterpret_cast<const unsigned*>(vp + 2 * e2);
        const int e = 2 * e2;
        const int n = n0 + rl + e;
        const float den0 = rowdot[rl + e] * inv_qk + 16.0f;
        const float den1 = rowdot[rl + e + 1] * inv_qk + 16.0f;
        out[(size_t)n * HCH + d] =
            __fdividef(acc[i][j][e] * inv_qk + 16.0f * unpk_lo(wv), den0);
        out[(size_t)(n + 1) * HCH + d] =
            __fdividef(acc[i][j][e + 1] * inv_qk + 16.0f * unpk_hi(wv), den1);
      }
    }
  }
}

// ---------------------------------------------------------------------------
extern "C" void kernel_launch(void* const* d_in, const int* in_sizes, int n_in,
                              void* d_out, int out_size, void* d_ws, size_t ws_size,
                              hipStream_t stream) {
  const float* x  = (const float*)d_in[0];
  const float* Wq = (const float*)d_in[1];
  const float* Wk = (const float*)d_in[2];
  const float* Wv = (const float*)d_in[3];

  char* ws = (char*)d_ws;
  const size_t projBytes = (size_t)N_NODES * HCH * 2;  // 32 MB each
  __bf16* qbf = (__bf16*)(ws);                         // natural [n][m]
  __bf16* kt  = (__bf16*)(ws + projBytes);             // transposed [g][m][l]
  __bf16* vsw = (__bf16*)(ws + 2 * projBytes);         // fragment-major
  __bf16* wsw = (__bf16*)(ws + 3 * projBytes);         // fragment-major W
  float* sumsq = (float*)(ws + 3 * projBytes + (size_t)3 * CH * HCH * 2);
  float* ksum  = (float*)((char*)sumsq + 256);
  __bf16* kvsw = (__bf16*)((char*)ksum + (size_t)NGRAPH * HCH * 4);
  float* out   = (float*)d_out;

  zero_kernel<<<(2112 + 255) / 256, 256, 0, stream>>>(sumsq, 2112);
  wt_kernel<<<(3 * CH * HCH + 255) / 256, 256, 0, stream>>>(Wq, Wk, Wv, wsw);
  proj_kernel<<<dim3(N_NODES / 128, 3), 256, 0, stream>>>(
      x, wsw, qbf, kt, vsw, sumsq, ksum);
  kv_kernel<<<NGRAPH, 256, 0, stream>>>(kt, vsw, kvsw);
  out_kernel<<<N_NODES / 128, 256, 0, stream>>>(qbf, vsw, kvsw, ksum, sumsq, out);
}
